// ACVRP_Decoder_2310692405752
// MI455X (gfx1250) — matrix operand order repaired and checked by reading, not compiled, not run on hardware
//
#include <hip/hip_runtime.h>
#include <hip/hip_bf16.h>
#include <math.h>

#define BB    64
#define NNODE 512
#define NQ    512
#define EE    128
#define HH    8
#define DD    16

typedef __attribute__((ext_vector_type(16))) _Float16 v16h;
typedef __attribute__((ext_vector_type(8)))  _Float16 v8h;
typedef __attribute__((ext_vector_type(8)))  float    v8f;

__device__ __forceinline__ v8f wmma_f16(v16h a, v16h b, v8f c) {
  // D = A(16x32 f16) * B(32x16 f16) + C(16x16 f32)
  return __builtin_amdgcn_wmma_f32_16x16x32_f16(false, a, false, b, (short)0, c,
                                                false, false);
}

// Concatenate two v8h into the 16x32 A-fragment / contiguous B-column slice.
__device__ __forceinline__ v16h cat8(v8h lo, v8h hi) {
  return __builtin_shufflevector(lo, hi, 0, 1, 2, 3, 4, 5, 6, 7,
                                 8, 9, 10, 11, 12, 13, 14, 15);
}

// Load a 16x32 A-fragment slice from an f16 row pointer (32 contiguous f16):
// elements {half*16 .. half*16+15} as two 16B loads (contiguous k order).
__device__ __forceinline__ v16h loadA16(const _Float16* xr, int half) {
  const v8h* p = (const v8h*)xr;
  return cat8(p[2 * half], p[2 * half + 1]);
}

__device__ __forceinline__ float redmax16(float v) {
#pragma unroll
  for (int off = 1; off < 16; off <<= 1) v = fmaxf(v, __shfl_xor(v, off, 32));
  return v;
}
__device__ __forceinline__ float redsum16(float v) {
#pragma unroll
  for (int off = 1; off < 16; off <<= 1) v += __shfl_xor(v, off, 32);
  return v;
}

// ---------------- f32 -> f16 elementwise convert ---------------------------
__global__ void cvt_f32_f16_kernel(const float* __restrict__ src,
                                   _Float16* __restrict__ dst, int n) {
  int i = blockIdx.x * blockDim.x + threadIdx.x;
  int stride = gridDim.x * blockDim.x;
  for (; i < n; i += stride) dst[i] = (_Float16)src[i];
}

// ---------------- Projections: k, v, q (= q1 + q0@Wq0^T) --------------------
// 1 wave computes a 16-row x 128-col tile. Output in head layout [B,H,n,D] f16.
__global__ void proj_kernel(const _Float16* __restrict__ enc16,
                            const _Float16* __restrict__ fr16,
                            const _Float16* __restrict__ q016,
                            const _Float16* __restrict__ Wk16,
                            const _Float16* __restrict__ Wv16,
                            const _Float16* __restrict__ Wq016,
                            const _Float16* __restrict__ Wq116,
                            _Float16* __restrict__ kh,
                            _Float16* __restrict__ vh,
                            _Float16* __restrict__ qh) {
  const int lane = threadIdx.x;
  const int half = lane >> 4;
  const int lr   = lane & 15;
  const int r0   = blockIdx.x * 16;   // row tile
  const int b    = blockIdx.y;        // batch
  const int which = blockIdx.z;       // 0=k, 1=v, 2=q

  const _Float16* X1;
  const _Float16* W1;
  const _Float16* X2 = nullptr;
  const _Float16* W2 = nullptr;
  _Float16* dst;
  if (which == 0)      { X1 = enc16 + (size_t)b * NNODE * EE; W1 = Wk16;  dst = kh; }
  else if (which == 1) { X1 = enc16 + (size_t)b * NNODE * EE; W1 = Wv16;  dst = vh; }
  else                 { X1 = q016  + (size_t)b * NQ * EE;    W1 = Wq016;
                         X2 = fr16 + (size_t)b * NQ * EE;     W2 = Wq116; dst = qh; }

#pragma unroll
  for (int t = 0; t < 8; ++t) {       // output col tile == head index
    v8f c = {0.f, 0.f, 0.f, 0.f, 0.f, 0.f, 0.f, 0.f};
#pragma unroll
    for (int k0 = 0; k0 < EE; k0 += 32) {
      v16h a  = loadA16(X1 + (size_t)(r0 + lr) * EE + k0, half);
      v16h bb = *(const v16h*)(W1 + (size_t)(t * 16 + lr) * EE + k0 + half * 16);
      c = wmma_f16(a, bb, c);
    }
    if (X2) {
#pragma unroll
      for (int k0 = 0; k0 < EE; k0 += 32) {
        v16h a  = loadA16(X2 + (size_t)(r0 + lr) * EE + k0, half);
        v16h bb = *(const v16h*)(W2 + (size_t)(t * 16 + lr) * EE + k0 + half * 16);
        c = wmma_f16(a, bb, c);
      }
    }
#pragma unroll
    for (int r = 0; r < 8; ++r) {
      int row = r0 + r + half * 8;
      dst[((size_t)(b * HH + t) * NQ + row) * DD + lr] = (_Float16)c[r];
    }
  }
}

// ---------------- Attention: 4 waves/block share LDS-staged K,V^T -----------
// Each wave owns one 16-row q tile; flash softmax over 32-key chunks:
// 2 half-K WMMAs for QK^T, 1 full-K WMMA for P*V (all operands from LDS).
__global__ void __launch_bounds__(128) attn_kernel(
    const _Float16* __restrict__ qh,
    const _Float16* __restrict__ kh,
    const _Float16* __restrict__ vh,
    const float* __restrict__ mask,
    _Float16* __restrict__ attn) {
  __shared__ alignas(32) _Float16 lk[NNODE][DD];        // 16 KB, row-major K
  __shared__ alignas(32) _Float16 lvT[DD][NNODE + 8];   // 16.25 KB, transposed V
  __shared__ alignas(32) _Float16 lp[4][16][32];        // 4 KB, per-wave P bounce
  const int tid  = threadIdx.x;
  const int wid  = tid >> 5;
  const int lane = tid & 31;
  const int half = lane >> 4;
  const int lr   = lane & 15;
  const int qr0 = blockIdx.x * 64 + wid * 16;
  const int h   = blockIdx.y;
  const int b   = blockIdx.z;

  const _Float16* kb = kh + (size_t)(b * HH + h) * NNODE * DD;
  const _Float16* vb = vh + (size_t)(b * HH + h) * NNODE * DD;

  // Cooperative staging: K rows as-is, V transposed (column reads -> b128).
  for (int row = tid; row < NNODE; row += 128) {
    v16h kv = *(const v16h*)(kb + (size_t)row * DD);
    v16h vv = *(const v16h*)(vb + (size_t)row * DD);
    *(v16h*)&lk[row][0] = kv;
#pragma unroll
    for (int d = 0; d < DD; ++d) lvT[d][row] = vv[d];
  }
  __syncthreads();

  // Q fragment (contract over D=16, upper K half zero)
  v16h aq;
  {
    v16h qv = *(const v16h*)(qh + ((size_t)(b * HH + h) * NQ + (qr0 + lr)) * DD);
#pragma unroll
    for (int i = 0; i < 16; ++i) aq[i] = (half == 0) ? qv[i] : (_Float16)0.f;
  }

  const float* mrow = mask + ((size_t)b * NQ + qr0) * NNODE;

  float m[8], l[8];
  v8f o = {0.f, 0.f, 0.f, 0.f, 0.f, 0.f, 0.f, 0.f};
#pragma unroll
  for (int r = 0; r < 8; ++r) { m[r] = -INFINITY; l[r] = 0.f; }
  const float rscale = 0.25f;  // 1/SQRT_QKV
  const v16h vzero = {};
  const v8f  fzero = {0.f, 0.f, 0.f, 0.f, 0.f, 0.f, 0.f, 0.f};

  for (int m0 = 0; m0 < NNODE; m0 += 32) {
    // Two K fragments from LDS: B[d][key] = K[key][d]; contract padded 16->32
    v16h bk0 = (half == 0) ? *(const v16h*)&lk[m0 + lr][0]      : vzero;
    v16h bk1 = (half == 0) ? *(const v16h*)&lk[m0 + 16 + lr][0] : vzero;
    v8f s0 = wmma_f16(aq, bk0, fzero);
    v8f s1 = wmma_f16(aq, bk1, fzero);

    float pv0[8], pv1[8];
#pragma unroll
    for (int r = 0; r < 8; ++r) {
      int row = r + half * 8;
      float sv0 = s0[r] * rscale + mrow[(size_t)row * NNODE + m0 + lr];
      float sv1 = s1[r] * rscale + mrow[(size_t)row * NNODE + m0 + 16 + lr];
      float mnew = fmaxf(m[r], redmax16(fmaxf(sv0, sv1)));
      float p0 = __expf(sv0 - mnew);
      float p1 = __expf(sv1 - mnew);
      float ts = redsum16(p0 + p1);
      float alpha = __expf(m[r] - mnew);
      l[r] = l[r] * alpha + ts;
      m[r] = mnew;
      o[r] *= alpha;
      pv0[r] = p0;
      pv1[r] = p1;
    }

    // P (16x32): C-layout -> LDS -> A-layout (full contract K=32)
    __syncthreads();
#pragma unroll
    for (int r = 0; r < 8; ++r) {
      lp[wid][r + half * 8][lr]      = (_Float16)pv0[r];
      lp[wid][r + half * 8][16 + lr] = (_Float16)pv1[r];
    }
    __syncthreads();
    v16h ap = loadA16(&lp[wid][lr][0], half);

    // V fragment from transposed LDS: B[key][d], K = half*16 + i, n = d = lr
    const v8h* pv = (const v8h*)&lvT[lr][m0 + half * 16];
    v16h bv = cat8(pv[0], pv[1]);

    o = wmma_f16(ap, bv, o);
  }

#pragma unroll
  for (int r = 0; r < 8; ++r) {
    int row = qr0 + r + half * 8;
    attn[((size_t)b * NQ + row) * (HH * DD) + h * DD + lr] = (_Float16)(o[r] / l[r]);
  }
}

// ---------------- mh = attn @ Wc^T + bc, stored f16 -------------------------
__global__ void out_proj_kernel(const _Float16* __restrict__ attn,
                                const _Float16* __restrict__ Wc16,
                                const float* __restrict__ bc,
                                _Float16* __restrict__ mh16) {
  const int lane = threadIdx.x;
  const int half = lane >> 4;
  const int lr   = lane & 15;
  const int r0 = blockIdx.x * 16;
  const int b  = blockIdx.y;
  const _Float16* X = attn + (size_t)b * NQ * (HH * DD);

#pragma unroll
  for (int t = 0; t < 8; ++t) {
    v8f c = {0.f, 0.f, 0.f, 0.f, 0.f, 0.f, 0.f, 0.f};
#pragma unroll
    for (int k0 = 0; k0 < HH * DD; k0 += 32) {
      v16h a  = loadA16(X + (size_t)(r0 + lr) * (HH * DD) + k0, half);
      v16h bb = *(const v16h*)(Wc16 + (size_t)(t * 16 + lr) * (HH * DD) + k0 + half * 16);
      c = wmma_f16(a, bb, c);
    }
    float bias = bc[t * 16 + lr];
#pragma unroll
    for (int r = 0; r < 8; ++r) {
      int row = r0 + r + half * 8;
      mh16[((size_t)b * NQ + row) * EE + t * 16 + lr] = (_Float16)(c[r] + bias);
    }
  }
}

// ---------------- pointer scores + tanh-clip + softmax ----------------------
__global__ void pointer_kernel(const _Float16* __restrict__ mh16,
                               const _Float16* __restrict__ enc16,
                               const float* __restrict__ mask,
                               float* __restrict__ out) {
  __shared__ alignas(32) float srow[16][NNODE];   // 32 KB
  const int lane = threadIdx.x;
  const int half = lane >> 4;
  const int lr   = lane & 15;
  const int qr0 = blockIdx.x * 16;
  const int b   = blockIdx.y;
  const _Float16* Xb = mh16 + (size_t)b * NQ * EE;
  const _Float16* Eb = enc16 + (size_t)b * NNODE * EE;

  for (int n0 = 0; n0 < NNODE; n0 += 16) {
    if (n0 + 16 < NNODE)
      __builtin_prefetch(Eb + (size_t)(n0 + 16 + lr) * EE, 0, 1);
    v8f c = {0.f, 0.f, 0.f, 0.f, 0.f, 0.f, 0.f, 0.f};
#pragma unroll
    for (int k0 = 0; k0 < EE; k0 += 32) {
      v16h a  = loadA16(Xb + (size_t)(qr0 + lr) * EE + k0, half);
      v16h bb = *(const v16h*)(Eb + (size_t)(n0 + lr) * EE + k0 + half * 16);
      c = wmma_f16(a, bb, c);
    }
#pragma unroll
    for (int r = 0; r < 8; ++r) srow[r + half * 8][n0 + lr] = c[r];
  }
  __syncthreads();

  const float inv_sqrtE = 1.0f / 11.313708498984761f;
  const float* mrow = mask + ((size_t)b * NQ + qr0) * NNODE;
  float* orow = out + ((size_t)b * NQ + qr0) * NNODE;

  for (int r = 0; r < 16; ++r) {
    float mx = -INFINITY;
#pragma unroll
    for (int j = 0; j < 16; ++j) {
      int col = lane + 32 * j;
      float sv = 10.0f * tanhf(srow[r][col] * inv_sqrtE) + mrow[(size_t)r * NNODE + col];
      srow[r][col] = sv;
      mx = fmaxf(mx, sv);
    }
#pragma unroll
    for (int off = 1; off < 32; off <<= 1) mx = fmaxf(mx, __shfl_xor(mx, off, 32));
    float sum = 0.f;
    float ev[16];
#pragma unroll
    for (int j = 0; j < 16; ++j) {
      int col = lane + 32 * j;
      ev[j] = __expf(srow[r][col] - mx);
      sum += ev[j];
    }
#pragma unroll
    for (int off = 1; off < 32; off <<= 1) sum += __shfl_xor(sum, off, 32);
    float inv = 1.f / sum;
#pragma unroll
    for (int j = 0; j < 16; ++j) orow[(size_t)r * NNODE + lane + 32 * j] = ev[j] * inv;
  }
}

extern "C" void kernel_launch(void* const* d_in, const int* in_sizes, int n_in,
                              void* d_out, int out_size, void* d_ws, size_t ws_size,
                              hipStream_t stream) {
  const float* enc       = (const float*)d_in[0];
  const float* first_row = (const float*)d_in[1];
  const float* q0        = (const float*)d_in[2];
  const float* mask      = (const float*)d_in[3];
  const float* Wq0       = (const float*)d_in[4];
  const float* Wq1       = (const float*)d_in[5];
  const float* Wk        = (const float*)d_in[6];
  const float* Wv        = (const float*)d_in[7];
  const float* Wc        = (const float*)d_in[8];
  const float* bc        = (const float*)d_in[9];
  float* out = (float*)d_out;

  const size_t MAT = (size_t)BB * NQ * EE;   // 4,194,304 elems
  const size_t WEL = (size_t)EE * EE;        // 16,384 elems per weight
  char* ws = (char*)d_ws;
  _Float16* kh    = (_Float16*)(ws + 0 * MAT * 2);
  _Float16* vh    = (_Float16*)(ws + 1 * MAT * 2);
  _Float16* qh    = (_Float16*)(ws + 2 * MAT * 2);
  _Float16* attn  = (_Float16*)(ws + 3 * MAT * 2);
  _Float16* mh16  = (_Float16*)(ws + 4 * MAT * 2);
  _Float16* enc16 = (_Float16*)(ws + 5 * MAT * 2);
  _Float16* fr16  = (_Float16*)(ws + 6 * MAT * 2);
  _Float16* q016  = (_Float16*)(ws + 7 * MAT * 2);
  _Float16* wbase = (_Float16*)(ws + 8 * MAT * 2);
  _Float16* Wq016 = wbase + 0 * WEL;
  _Float16* Wq116 = wbase + 1 * WEL;
  _Float16* Wk16  = wbase + 2 * WEL;
  _Float16* Wv16  = wbase + 3 * WEL;
  _Float16* Wc16  = wbase + 4 * WEL;

  // One-time f16 conversions (activations + weights)
  cvt_f32_f16_kernel<<<4096, 256, 0, stream>>>(enc,       enc16, (int)MAT);
  cvt_f32_f16_kernel<<<4096, 256, 0, stream>>>(first_row, fr16,  (int)MAT);
  cvt_f32_f16_kernel<<<4096, 256, 0, stream>>>(q0,        q016,  (int)MAT);
  cvt_f32_f16_kernel<<<64, 256, 0, stream>>>(Wq0, Wq016, (int)WEL);
  cvt_f32_f16_kernel<<<64, 256, 0, stream>>>(Wq1, Wq116, (int)WEL);
  cvt_f32_f16_kernel<<<64, 256, 0, stream>>>(Wk,  Wk16,  (int)WEL);
  cvt_f32_f16_kernel<<<64, 256, 0, stream>>>(Wv,  Wv16,  (int)WEL);
  cvt_f32_f16_kernel<<<64, 256, 0, stream>>>(Wc,  Wc16,  (int)WEL);

  // k, v, q projections (f16 WMMA, f32 accumulate)
  proj_kernel<<<dim3(NQ / 16, BB, 3), 32, 0, stream>>>(
      enc16, fr16, q016, Wk16, Wv16, Wq016, Wq116, kh, vh, qh);

  // multi-head attention, 4 waves/block sharing LDS-staged K and V^T
  attn_kernel<<<dim3(NQ / 64, HH, BB), 128, 0, stream>>>(qh, kh, vh, mask, attn);

  // output projection + bias
  out_proj_kernel<<<dim3(NQ / 16, BB), 32, 0, stream>>>(attn, Wc16, bc, mh16);

  // pointer scoring + tanh clip + mask + softmax
  pointer_kernel<<<dim3(NQ / 16, BB), 32, 0, stream>>>(mh16, enc16, mask, out);
}
